// ChebyKANProjector_6433861010046
// MI455X (gfx1250) — matrix operand order repaired and checked by reading, not compiled, not run on hardware
//
#include <hip/hip_runtime.h>
#include <hip/hip_bf16.h>
#include <math.h>

// ---------------------------------------------------------------------------
// ChebyKAN 2-layer MLP for MI455X (gfx1250), bf16 WMMA path.
//   B=8192, IN=2048, H=2048, OUT=512, DEG=4 (5 Chebyshev basis fns)
// Both layers are one GEMM with augmented K = Cin*5:
//   columns per input feature i: [x, tanh(x), T2, T3, T4]; T0 folds into bias.
// Compute-bound (AI ~1300 FLOP/B) -> maximize WMMA density per LDS load:
//   64x64 per wave = 16 WMMAs per 16 ds_load_b128 per k-step.
// ---------------------------------------------------------------------------

typedef __attribute__((ext_vector_type(16))) __bf16 v16bf;
typedef __attribute__((ext_vector_type(8)))  __bf16 v8bf;
typedef __attribute__((ext_vector_type(8)))  float  v8f;

#define B_ROWS 8192
#define IN_F   2048
#define H_F    2048
#define OUT_F  512
#define DEGP1  5
#define BN_EPS 1e-5f

#define MTILE 256         // output tile rows per block
#define NTILE 128         // output tile cols per block
#define KC 160            // K-chunk = 32 input features * 5 basis columns
#define LDSK 160          // LDS row stride (elements); 320B rows, 32B aligned

// ---------------------------------------------------------------------------
// Weight packing: Wp[k*N + n], k = i*5 + c, c==0 -> base weight, c>=1 -> s*c
// ---------------------------------------------------------------------------
__global__ __launch_bounds__(256)
void pack_w1_kernel(const float* __restrict__ w1, const float* __restrict__ c1,
                    const float* __restrict__ s1, __bf16* __restrict__ Wp) {
    size_t idx = (size_t)blockIdx.x * 256 + threadIdx.x;
    int k = (int)(idx / H_F);
    int h = (int)(idx % H_F);
    int i = k / DEGP1, c = k % DEGP1;
    float v = (c == 0) ? w1[(size_t)h * IN_F + i]
                       : s1[h] * c1[((size_t)i * H_F + h) * DEGP1 + c];
    Wp[idx] = (__bf16)v;
}

__global__ __launch_bounds__(256)
void pack_w2_kernel(const float* __restrict__ w2, const float* __restrict__ c2,
                    const float* __restrict__ s2, __bf16* __restrict__ Wp) {
    size_t idx = (size_t)blockIdx.x * 256 + threadIdx.x;
    int k = (int)(idx / OUT_F);
    int o = (int)(idx % OUT_F);
    int i = k / DEGP1, c = k % DEGP1;
    float v = (c == 0) ? w2[(size_t)o * H_F + i]
                       : s2[o] * c2[((size_t)i * OUT_F + o) * DEGP1 + c];
    Wp[idx] = (__bf16)v;
}

// bias_eff[n] = b[n] + s[n] * sum_i coeff[i, n, 0]    (T0 == 1 contribution)
__global__ __launch_bounds__(256)
void bias1_kernel(const float* __restrict__ b1, const float* __restrict__ s1,
                  const float* __restrict__ c1, float* __restrict__ bias) {
    int h = blockIdx.x * 256 + threadIdx.x;
    if (h >= H_F) return;
    float s = 0.f;
    for (int i = 0; i < IN_F; ++i) s += c1[((size_t)i * H_F + h) * DEGP1];
    bias[h] = b1[h] + s1[h] * s;
}

__global__ __launch_bounds__(256)
void bias2_kernel(const float* __restrict__ b2, const float* __restrict__ s2,
                  const float* __restrict__ c2, float* __restrict__ bias) {
    int o = blockIdx.x * 256 + threadIdx.x;
    if (o >= OUT_F) return;
    float s = 0.f;
    for (int i = 0; i < H_F; ++i) s += c2[((size_t)i * OUT_F + o) * DEGP1];
    bias[o] = b2[o] + s2[o] * s;
}

// Training-mode BN folded to affine: a = gamma*rsqrt(var+eps), b = beta - a*mu
__global__ __launch_bounds__(256)
void bn_stats_kernel(const float* __restrict__ hbuf,
                     const float* __restrict__ gamma,
                     const float* __restrict__ beta,
                     float* __restrict__ a, float* __restrict__ b) {
    int c = blockIdx.x * 256 + threadIdx.x;
    if (c >= H_F) return;
    float s = 0.f, sq = 0.f;
    for (int r = 0; r < B_ROWS; ++r) {
        float v = hbuf[(size_t)r * H_F + c];
        s += v; sq += v * v;
    }
    float mean = s * (1.f / B_ROWS);
    float var  = sq * (1.f / B_ROWS) - mean * mean;
    float sc   = gamma[c] * rsqrtf(var + BN_EPS);
    a[c] = sc;
    b[c] = beta[c] - sc * mean;
}

// ---------------------------------------------------------------------------
// Fused ChebyKAN GEMM: Out[m,n] = bias[n] + sum_k A(x)[m,k] * Wp[k,n]
// Activation tile generated on the fly into LDS (tanh + Chebyshev recurrence),
// optional fused BN affine on the input.
// 256 threads = 8 wave32s in a 4x2 grid; wave tile = 64x64 (4x4 WMMA tiles).
// ---------------------------------------------------------------------------
__global__ __launch_bounds__(256)
void cheby_gemm_kernel(const float* __restrict__ X, int Cin,
                       const __bf16* __restrict__ Wp,
                       const float* __restrict__ bias,
                       const float* __restrict__ aff_a,
                       const float* __restrict__ aff_b, int use_aff,
                       float* __restrict__ Out, int Nn) {
    __shared__ __align__(32) __bf16 Atile[MTILE][LDSK];   // [m][k], 80 KB
    __shared__ __align__(32) __bf16 Btile[NTILE][LDSK];   // [n][k], 40 KB

    const int tid  = threadIdx.x;
    const int lane = tid & 31;
    const int wave = tid >> 5;
    const int wr   = wave >> 1;      // 0..3 : wave row (64-row bands)
    const int wc   = wave & 1;       // 0..1 : wave col (64-col bands)
    const int lm   = lane & 15;
    const int lh   = lane >> 4;

    const int m0 = blockIdx.y * MTILE;
    const int n0 = blockIdx.x * NTILE;

    v8f acc[4][4];
#pragma unroll
    for (int mt = 0; mt < 4; ++mt)
#pragma unroll
        for (int nt = 0; nt < 4; ++nt) acc[mt][nt] = {};

    const int K       = Cin * DEGP1;
    const int kchunks = K / KC;

    for (int kc = 0; kc < kchunks; ++kc) {
        const int i0 = kc * 32;          // 32 input features per chunk
        // ---- stage A: generate bf16 activation tile [256 x 160] ----
        for (int e = tid; e < MTILE * 32; e += 256) {
            const int m  = e >> 5;
            const int ii = e & 31;
            float v = X[(size_t)(m0 + m) * Cin + (i0 + ii)];
            if (use_aff) v = aff_a[i0 + ii] * v + aff_b[i0 + ii];
            float xn = tanhf(v);
            float t2 = 2.f * xn * xn - 1.f;
            float t3 = 2.f * xn * t2 - xn;
            float t4 = 2.f * xn * t3 - t2;
            __bf16* dst = &Atile[m][ii * DEGP1];
            dst[0] = (__bf16)v;  dst[1] = (__bf16)xn; dst[2] = (__bf16)t2;
            dst[3] = (__bf16)t3; dst[4] = (__bf16)t4;
        }
        // ---- stage B: packed weights, stored N-major [128 x 160] ----
        const int kb = kc * KC;
        for (int e = tid; e < KC * NTILE; e += 256) {
            const int kk = e >> 7;       // 0..159
            const int n  = e & 127;
            Btile[n][kk] = Wp[(size_t)(kb + kk) * Nn + (n0 + n)];
            if (kc + 1 < kchunks)        // stream next weight chunk into L2
                __builtin_prefetch(&Wp[(size_t)(kb + KC + kk) * Nn + (n0 + n)], 0, 1);
        }
        __syncthreads();

#pragma unroll
        for (int ks = 0; ks < KC / 32; ++ks) {
            const int ko = ks * 32;
            // A fragments: lanes 0-15 hold K{0-15}, lanes 16-31 K{16-31}
            v16bf afrag[4];
#pragma unroll
            for (int mt = 0; mt < 4; ++mt) {
                const __bf16* ar = &Atile[wr * 64 + mt * 16 + lm][0];
                v8bf lo = *(const v8bf*)(ar + ko + lh * 16);
                v8bf hi = *(const v8bf*)(ar + ko + lh * 16 + 8);
                afrag[mt] = __builtin_shufflevector(lo, hi, 0, 1, 2, 3, 4, 5, 6, 7,
                                                    8, 9, 10, 11, 12, 13, 14, 15);
            }
            // B fragments: N = lane%16, K = ko + (lane/16)*16 .. +16, contiguous
            v16bf bfrag[4];
#pragma unroll
            for (int nt = 0; nt < 4; ++nt)
                bfrag[nt] =
                    *(const v16bf*)(&Btile[wc * 64 + nt * 16 + lm][ko + lh * 16]);
#pragma unroll
            for (int mt = 0; mt < 4; ++mt)
#pragma unroll
                for (int nt = 0; nt < 4; ++nt)
                    acc[mt][nt] = __builtin_amdgcn_wmma_f32_16x16x32_bf16(
                        false, afrag[mt], false, bfrag[nt], (short)0, acc[mt][nt],
                        false, false);
        }
        __syncthreads();
    }

    // ---- epilogue: C layout VGPR r -> M = r + (lane/16)*8, N = lane%16 ----
#pragma unroll
    for (int mt = 0; mt < 4; ++mt) {
#pragma unroll
        for (int nt = 0; nt < 4; ++nt) {
            const int col = n0 + wc * 64 + nt * 16 + lm;
            const float bv = bias[col];
#pragma unroll
            for (int r = 0; r < 8; ++r) {
                const int row = m0 + wr * 64 + mt * 16 + r + lh * 8;
                Out[(size_t)row * Nn + col] = acc[mt][nt][r] + bv;
            }
        }
    }
}

// ---------------------------------------------------------------------------
extern "C" void kernel_launch(void* const* d_in, const int* in_sizes, int n_in,
                              void* d_out, int out_size, void* d_ws, size_t ws_size,
                              hipStream_t stream) {
    (void)in_sizes; (void)n_in; (void)out_size; (void)ws_size;

    const float* x      = (const float*)d_in[0];
    const float* w1     = (const float*)d_in[1];
    const float* b1     = (const float*)d_in[2];
    const float* c1     = (const float*)d_in[3];
    const float* s1     = (const float*)d_in[4];
    const float* gamma  = (const float*)d_in[5];
    const float* beta   = (const float*)d_in[6];
    const float* w2     = (const float*)d_in[7];
    const float* b2     = (const float*)d_in[8];
    const float* c2     = (const float*)d_in[9];
    const float* s2     = (const float*)d_in[10];
    float* out = (float*)d_out;

    // workspace layout (256B aligned)
    char* ws = (char*)d_ws;
    const size_t w1p_bytes = (size_t)IN_F * DEGP1 * H_F * sizeof(__bf16);   // 40 MB
    const size_t w2p_bytes = (size_t)H_F * DEGP1 * OUT_F * sizeof(__bf16);  // 10 MB
    const size_t h_bytes   = (size_t)B_ROWS * H_F * sizeof(float);          // 64 MB
    size_t off = 0;
    __bf16* W1p  = (__bf16*)(ws + off); off += (w1p_bytes + 255) & ~255ull;
    __bf16* W2p  = (__bf16*)(ws + off); off += (w2p_bytes + 255) & ~255ull;
    float* hbuf  = (float*)(ws + off);  off += (h_bytes + 255) & ~255ull;
    float* bias1 = (float*)(ws + off);  off += (H_F * sizeof(float) + 255) & ~255ull;
    float* bias2 = (float*)(ws + off);  off += (OUT_F * sizeof(float) + 255) & ~255ull;
    float* bn_a  = (float*)(ws + off);  off += (H_F * sizeof(float) + 255) & ~255ull;
    float* bn_b  = (float*)(ws + off);

    // 1) pack weights + effective biases
    pack_w1_kernel<<<(IN_F * DEGP1 * H_F) / 256, 256, 0, stream>>>(w1, c1, s1, W1p);
    pack_w2_kernel<<<(H_F * DEGP1 * OUT_F) / 256, 256, 0, stream>>>(w2, c2, s2, W2p);
    bias1_kernel<<<(H_F + 255) / 256, 256, 0, stream>>>(b1, s1, c1, bias1);
    bias2_kernel<<<(OUT_F + 255) / 256, 256, 0, stream>>>(b2, s2, c2, bias2);

    // 2) layer 1: h = ChebyKAN(x)           [8192 x 2048]
    dim3 g1(H_F / NTILE, B_ROWS / MTILE);
    cheby_gemm_kernel<<<g1, 256, 0, stream>>>(x, IN_F, W1p, bias1,
                                              nullptr, nullptr, 0, hbuf, H_F);

    // 3) batch-norm statistics -> affine
    bn_stats_kernel<<<(H_F + 255) / 256, 256, 0, stream>>>(hbuf, gamma, beta,
                                                           bn_a, bn_b);

    // 4) layer 2: out = ChebyKAN(BN(h))     [8192 x 512]
    dim3 g2(OUT_F / NTILE, B_ROWS / MTILE);
    cheby_gemm_kernel<<<g2, 256, 0, stream>>>(hbuf, H_F, W2p, bias2,
                                              bn_a, bn_b, 1, out, OUT_F);
}
